// BatchInvariantMultiHeadAttention_33303176413803
// MI455X (gfx1250) — matrix operand order repaired and checked by reading, not compiled, not run on hardware
//
#include <hip/hip_runtime.h>

// ---------------------------------------------------------------------------
// BatchInvariantMultiHeadAttention for MI455X (gfx1250, wave32, WMMA bf16)
//   pass 0: f32 -> bf16 conversion of activations + weights (bandwidth bound)
//   pass 1: fused QKV projection GEMM (bf16 WMMA, f32 accum)
//   pass 2: causal flash attention (bf16 WMMA, f32 online softmax)
//   pass 3: output projection GEMM (bf16 WMMA, f32 out)
// ---------------------------------------------------------------------------

typedef __bf16 bf16_t;
typedef __attribute__((ext_vector_type(16))) __bf16 v16bf;
typedef __attribute__((ext_vector_type(8)))  __bf16 v8bf;
typedef __attribute__((ext_vector_type(8)))  float  v8f;

// problem dims (fixed by the reference)
constexpr int S_ = 2048, B_ = 2, E_ = 1024, H_ = 16, D_ = 64;
constexpr int M_ = S_ * B_;  // 4096 GEMM rows

#define WMMA_BF16(a, b, c) \
  __builtin_amdgcn_wmma_f32_16x16x32_bf16(false, (a), false, (b), (short)0, (c), false, false)

// ----- fragment loaders (layouts per CDNA5 ISA 7.12.2) ----------------------
// A 16x32 bf16: lane m = l&15, half h = l>>4; elems 0..7 -> K=h*8+i, 8..15 -> K=16+h*8+i
__device__ __forceinline__ v16bf a_frag_bf16(const bf16_t* __restrict__ base, int ld, int k0) {
  const int lane = threadIdx.x & 31;
  const int m = lane & 15, h = lane >> 4;
  const bf16_t* p = base + (size_t)m * ld + k0 + h * 8;
  v8bf lo = *(const v8bf*)p;
  v8bf hi = *(const v8bf*)(p + 16);
  v16bf a;
#pragma unroll
  for (int i = 0; i < 8; ++i) { a[i] = lo[i]; a[8 + i] = hi[i]; }
  return a;
}

// B 32x16 bf16, column-contiguous source: elem (k,n) = src[n*ldn + k]
// (also serves W.T fragments from row-major bf16 W[N,K])
__device__ __forceinline__ v16bf b_frag_bf16(const bf16_t* __restrict__ src, int ldn, int n0, int k0) {
  const int lane = threadIdx.x & 31;
  const int n = n0 + (lane & 15), h = lane >> 4;
  const bf16_t* p = src + (size_t)n * ldn + k0 + h * 8;
  v8bf lo = *(const v8bf*)p;
  v8bf hi = *(const v8bf*)(p + 16);
  v16bf b;
#pragma unroll
  for (int i = 0; i < 8; ++i) { b[i] = lo[i]; b[8 + i] = hi[i]; }
  return b;
}

// ---------------------------------------------------------------------------
// Pass 0: f32 -> bf16 (8 elements / thread, vectorized b128 in / b128 out)
// ---------------------------------------------------------------------------
__global__ __launch_bounds__(256) void cvt_f32_bf16_kernel(
    const float* __restrict__ src, bf16_t* __restrict__ dst, int n8) {
  const int idx = blockIdx.x * 256 + threadIdx.x;
  if (idx >= n8) return;
  const size_t base = (size_t)idx * 8;
  v8bf o;
#pragma unroll
  for (int i = 0; i < 8; ++i) o[i] = (bf16_t)src[base + i];
  *(v8bf*)(dst + base) = o;
}

// ---------------------------------------------------------------------------
// Pass 1: fused QKV projection. Y = X @ W.T + b (bf16 in, bf16 out)
// which=0: Q -> [B,H,S,D]; which=1: K -> [B,H,S,D]; which=2: V -> [B,H,D,S]
// block tile 64x128, 4 waves, each wave 32x64 (8 WMMAs / K-step)
// ---------------------------------------------------------------------------
__global__ __launch_bounds__(128) void qkv_proj_kernel(
    const bf16_t* __restrict__ Xq, const bf16_t* __restrict__ Xk, const bf16_t* __restrict__ Xv,
    const bf16_t* __restrict__ Wq, const float* __restrict__ bq,
    const bf16_t* __restrict__ Wk, const float* __restrict__ bk,
    const bf16_t* __restrict__ Wv, const float* __restrict__ bv,
    bf16_t* __restrict__ qout, bf16_t* __restrict__ kout, bf16_t* __restrict__ vtout) {
  const int which = blockIdx.z;
  const bf16_t* X    = (which == 0) ? Xq : (which == 1) ? Xk : Xv;
  const bf16_t* W    = (which == 0) ? Wq : (which == 1) ? Wk : Wv;
  const float*  bias = (which == 0) ? bq : (which == 1) ? bk : bv;

  const int wave = threadIdx.x >> 5;
  const int lane = threadIdx.x & 31;
  const int mbase = blockIdx.x * 64 + (wave >> 1) * 32;
  const int nbase = blockIdx.y * 128 + (wave & 1) * 64;

  v8f c[2][4] = {};
  const bf16_t* Arow = X + (size_t)mbase * E_;
  for (int k0 = 0; k0 < E_; k0 += 32) {
    // issue all 12 b128 loads of this K-step before any WMMA consumes them
    const v16bf a0 = a_frag_bf16(Arow, E_, k0);
    const v16bf a1 = a_frag_bf16(Arow + (size_t)16 * E_, E_, k0);
    v16bf bf[4];
#pragma unroll
    for (int ni = 0; ni < 4; ++ni) bf[ni] = b_frag_bf16(W, E_, nbase + ni * 16, k0);
    // software prefetch ~256B ahead on this wave's A rows and B columns
    if (k0 + 128 < E_) {
      __builtin_prefetch(Arow + (size_t)(lane & 15) * E_ + k0 + 128, 0, 3);
      __builtin_prefetch(Arow + (size_t)(16 + (lane & 15)) * E_ + k0 + 128, 0, 3);
      __builtin_prefetch(W + (size_t)(nbase + ((lane & 15) << 2)) * E_ + k0 + 128, 0, 3);
    }
#pragma unroll
    for (int ni = 0; ni < 4; ++ni) {
      c[0][ni] = WMMA_BF16(a0, bf[ni], c[0][ni]);
      c[1][ni] = WMMA_BF16(a1, bf[ni], c[1][ni]);
    }
  }

  const int h = lane >> 4, nl = lane & 15;
#pragma unroll
  for (int mi = 0; mi < 2; ++mi)
#pragma unroll
    for (int ni = 0; ni < 4; ++ni)
#pragma unroll
      for (int r = 0; r < 8; ++r) {
        const int row = mbase + mi * 16 + r + 8 * h;   // row = s*B + b
        const int col = nbase + ni * 16 + nl;          // col = hd*D + d
        const float y = c[mi][ni][r] + bias[col];
        const int s = row >> 1, b = row & 1;           // B_ == 2
        const int hd = col >> 6, d = col & 63;         // D_ == 64
        const bf16_t v = (bf16_t)y;
        if (which == 2)
          vtout[(((size_t)((b << 4) + hd)) * D_ + d) * S_ + s] = v;   // [B,H,D,S]
        else if (which == 0)
          qout[(((size_t)((b << 4) + hd)) * S_ + s) * D_ + d] = v;    // [B,H,S,D]
        else
          kout[(((size_t)((b << 4) + hd)) * S_ + s) * D_ + d] = v;
      }
}

// ---------------------------------------------------------------------------
// Pass 2: causal flash attention, one (b,h) x 64 query rows per block.
// 4 waves x 16 query rows. Online softmax in f32, P tile via LDS transpose.
// ---------------------------------------------------------------------------
__global__ __launch_bounds__(128) void attn_kernel(
    const bf16_t* __restrict__ qbf, const bf16_t* __restrict__ kbf,
    const bf16_t* __restrict__ vtb, bf16_t* __restrict__ obf) {
  __shared__ __align__(16) bf16_t pbuf[4][16][40];  // padded rows (80B) keep 16B alignment
  constexpr float scale = 0.125f;                   // 1/sqrt(D)

  const int wave = threadIdx.x >> 5;
  const int lane = threadIdx.x & 31;
  const int bh = blockIdx.y;
  const int b = bh >> 4, hd = bh & 15;              // H_ == 16
  const int q0 = blockIdx.x * 64 + wave * 16;       // this wave's 16 query rows

  const bf16_t* qp = qbf + ((size_t)bh * S_ + q0) * D_;
  const bf16_t* kp = kbf + (size_t)bh * S_ * D_;    // elem (d, s_k) at kp[s_k*D + d]
  const bf16_t* vp = vtb + (size_t)bh * D_ * S_;    // elem (s_k, d) at vp[d*S + s_k]

  const v16bf qa0 = a_frag_bf16(qp, D_, 0);
  const v16bf qa1 = a_frag_bf16(qp, D_, 32);

  v8f o[4] = {};
  float mrun[8], lrun[8];
#pragma unroll
  for (int r = 0; r < 8; ++r) { mrun[r] = -INFINITY; lrun[r] = 0.f; }

  const int h = lane >> 4, nl = lane & 15;
  const int nkb = ((q0 + 15) >> 5) + 1;             // key blocks of 32, causal limit

  for (int kb = 0; kb < nkb; ++kb) {
    const int kc = kb << 5;
    // prefetch next key block (K rows + V columns) while we compute this one
    if (kb + 1 < nkb) {
      const int kn = kc + 32;
      __builtin_prefetch(kp + (size_t)(kn + lane) * D_, 0, 3);            // next 32 K rows
      __builtin_prefetch(vp + (size_t)(lane << 1) * S_ + kn, 0, 3);       // next V cols
    }
    // ---- scores: S = Q @ K^T (16 x 32 tile, f32 accum) ----
    v8f sf[2];
#pragma unroll
    for (int ng = 0; ng < 2; ++ng) {
      const v16bf bk0 = b_frag_bf16(kp, D_, kc + ng * 16, 0);
      const v16bf bk1 = b_frag_bf16(kp, D_, kc + ng * 16, 32);
      v8f s = {};
      s = WMMA_BF16(qa0, bk0, s);
      s = WMMA_BF16(qa1, bk1, s);
      sf[ng] = s;
    }
    // scale + causal mask
#pragma unroll
    for (int ng = 0; ng < 2; ++ng)
#pragma unroll
      for (int r = 0; r < 8; ++r) {
        const int qg = q0 + r + 8 * h;
        const int kg = kc + ng * 16 + nl;
        const float sv = sf[ng][r] * scale;
        sf[ng][r] = (kg > qg) ? -INFINITY : sv;
      }
    // row max across the 32 keys (reduce over the 16 lanes of the half-wave)
    float mx[8];
#pragma unroll
    for (int r = 0; r < 8; ++r) mx[r] = fmaxf(sf[0][r], sf[1][r]);
#pragma unroll
    for (int off = 1; off < 16; off <<= 1)
#pragma unroll
      for (int r = 0; r < 8; ++r) mx[r] = fmaxf(mx[r], __shfl_xor(mx[r], off, 32));
    // online softmax update
    float alpha[8];
#pragma unroll
    for (int r = 0; r < 8; ++r) {
      const float mn = fmaxf(mrun[r], mx[r]);
      alpha[r] = __expf(mrun[r] - mn);
      mrun[r] = mn;
    }
    float rs[8];
#pragma unroll
    for (int r = 0; r < 8; ++r) rs[r] = 0.f;
#pragma unroll
    for (int ng = 0; ng < 2; ++ng)
#pragma unroll
      for (int r = 0; r < 8; ++r) {
        const float p = __expf(sf[ng][r] - mrun[r]);  // exp(-inf - m) == 0
        sf[ng][r] = p;
        rs[r] += p;
      }
#pragma unroll
    for (int off = 1; off < 16; off <<= 1)
#pragma unroll
      for (int r = 0; r < 8; ++r) rs[r] += __shfl_xor(rs[r], off, 32);
#pragma unroll
    for (int r = 0; r < 8; ++r) lrun[r] = lrun[r] * alpha[r] + rs[r];
#pragma unroll
    for (int j = 0; j < 4; ++j)
#pragma unroll
      for (int r = 0; r < 8; ++r) o[j][r] *= alpha[r];

    // ---- C-layout -> A-layout transpose of P through LDS (bf16) ----
#pragma unroll
    for (int ng = 0; ng < 2; ++ng)
#pragma unroll
      for (int r = 0; r < 8; ++r)
        pbuf[wave][r + 8 * h][ng * 16 + nl] = (bf16_t)sf[ng][r];
    asm volatile("s_wait_dscnt 0" ::: "memory");  // per-wave LDS ops are in-order

    v16bf pa;
    {
      const int m = lane & 15, hh = lane >> 4;
      v8bf lo = *(const v8bf*)&pbuf[wave][m][hh * 8];
      v8bf hi = *(const v8bf*)&pbuf[wave][m][16 + hh * 8];
#pragma unroll
      for (int i = 0; i < 8; ++i) { pa[i] = lo[i]; pa[8 + i] = hi[i]; }
    }
    // ---- O += P @ V (k-dim = 32 keys of this block) ----
#pragma unroll
    for (int j = 0; j < 4; ++j) {
      const v16bf bv = b_frag_bf16(vp, S_, j * 16, kc);
      o[j] = WMMA_BF16(pa, bv, o[j]);
    }
  }

  // normalize and store as bf16 in [S,B,E] layout for the output projection
  float inv[8];
#pragma unroll
  for (int r = 0; r < 8; ++r) inv[r] = 1.f / lrun[r];
#pragma unroll
  for (int j = 0; j < 4; ++j)
#pragma unroll
    for (int r = 0; r < 8; ++r) {
      const int qg = q0 + r + 8 * h;
      const int e = (hd << 6) + j * 16 + nl;
      obf[(((size_t)qg << 1) + b) * E_ + e] = (bf16_t)(o[j][r] * inv[r]);
    }
}

// ---------------------------------------------------------------------------
// Pass 3: output projection. out = A @ Wo.T + bo (bf16 in, f32 out)
// block tile 64x128, 4 waves, each wave 32x64 (8 WMMAs / K-step)
// ---------------------------------------------------------------------------
__global__ __launch_bounds__(128) void out_proj_kernel(
    const bf16_t* __restrict__ A, const bf16_t* __restrict__ Wo, const float* __restrict__ bo,
    float* __restrict__ out) {
  const int wave = threadIdx.x >> 5;
  const int lane = threadIdx.x & 31;
  const int mbase = blockIdx.x * 64 + (wave >> 1) * 32;
  const int nbase = blockIdx.y * 128 + (wave & 1) * 64;

  v8f c[2][4] = {};
  const bf16_t* Ab = A + (size_t)mbase * E_;
  for (int k0 = 0; k0 < E_; k0 += 32) {
    const v16bf a0 = a_frag_bf16(Ab, E_, k0);
    const v16bf a1 = a_frag_bf16(Ab + (size_t)16 * E_, E_, k0);
    v16bf bf[4];
#pragma unroll
    for (int ni = 0; ni < 4; ++ni) bf[ni] = b_frag_bf16(Wo, E_, nbase + ni * 16, k0);
    if (k0 + 128 < E_) {
      __builtin_prefetch(Ab + (size_t)(lane & 15) * E_ + k0 + 128, 0, 3);
      __builtin_prefetch(Ab + (size_t)(16 + (lane & 15)) * E_ + k0 + 128, 0, 3);
      __builtin_prefetch(Wo + (size_t)(nbase + ((lane & 15) << 2)) * E_ + k0 + 128, 0, 3);
    }
#pragma unroll
    for (int ni = 0; ni < 4; ++ni) {
      c[0][ni] = WMMA_BF16(a0, bf[ni], c[0][ni]);
      c[1][ni] = WMMA_BF16(a1, bf[ni], c[1][ni]);
    }
  }

  const int h = lane >> 4, nl = lane & 15;
#pragma unroll
  for (int mi = 0; mi < 2; ++mi)
#pragma unroll
    for (int ni = 0; ni < 4; ++ni)
#pragma unroll
      for (int r = 0; r < 8; ++r) {
        const int row = mbase + mi * 16 + r + 8 * h;
        const int col = nbase + ni * 16 + nl;
        out[(size_t)row * E_ + col] = c[mi][ni][r] + bo[col];
      }
}

// ---------------------------------------------------------------------------
extern "C" void kernel_launch(void* const* d_in, const int* in_sizes, int n_in,
                              void* d_out, int out_size, void* d_ws, size_t ws_size,
                              hipStream_t stream) {
  (void)in_sizes; (void)n_in; (void)out_size; (void)ws_size;

  const float* query = (const float*)d_in[0];
  const float* key_  = (const float*)d_in[1];
  const float* value = (const float*)d_in[2];
  const float* Wq = (const float*)d_in[3];  const float* bq = (const float*)d_in[4];
  const float* Wk = (const float*)d_in[5];  const float* bk = (const float*)d_in[6];
  const float* Wv = (const float*)d_in[7];  const float* bv = (const float*)d_in[8];
  const float* Wo = (const float*)d_in[9];  const float* bo = (const float*)d_in[10];
  float* out = (float*)d_out;

  // ---- workspace layout (bf16) ----
  const size_t nX = (size_t)M_ * E_;        // 4,194,304 per activation / head tensor
  const size_t nW = (size_t)E_ * E_;        // 1,048,576 per weight
  bf16_t* xq_bf = (bf16_t*)d_ws;            // [S*B, E]
  bf16_t* xk_bf = xq_bf + nX;
  bf16_t* xv_bf = xk_bf + nX;
  bf16_t* wq_bf = xv_bf + nX;               // [E, E]
  bf16_t* wk_bf = wq_bf + nW;
  bf16_t* wv_bf = wk_bf + nW;
  bf16_t* wo_bf = wv_bf + nW;
  bf16_t* q_bf  = wo_bf + nW;               // [B,H,S,D]
  bf16_t* k_bf  = q_bf + nX;                // [B,H,S,D]
  bf16_t* vT_bf = k_bf + nX;                // [B,H,D,S]
  bf16_t* o_bf  = vT_bf + nX;               // [S,B,E]

  // ---- pass 0: convert everything to bf16 once ----
  {
    const int nx8 = (int)(nX / 8), nw8 = (int)(nW / 8);
    const int gx = (nx8 + 255) / 256, gw = (nw8 + 255) / 256;
    cvt_f32_bf16_kernel<<<gx, 256, 0, stream>>>(query, xq_bf, nx8);
    cvt_f32_bf16_kernel<<<gx, 256, 0, stream>>>(key_,  xk_bf, nx8);
    cvt_f32_bf16_kernel<<<gx, 256, 0, stream>>>(value, xv_bf, nx8);
    cvt_f32_bf16_kernel<<<gw, 256, 0, stream>>>(Wq, wq_bf, nw8);
    cvt_f32_bf16_kernel<<<gw, 256, 0, stream>>>(Wk, wk_bf, nw8);
    cvt_f32_bf16_kernel<<<gw, 256, 0, stream>>>(Wv, wv_bf, nw8);
    cvt_f32_bf16_kernel<<<gw, 256, 0, stream>>>(Wo, wo_bf, nw8);
  }

  // ---- pass 1: QKV projections ----
  dim3 g1(M_ / 64, E_ / 128, 3);
  qkv_proj_kernel<<<g1, 128, 0, stream>>>(xq_bf, xk_bf, xv_bf,
                                          wq_bf, bq, wk_bf, bk, wv_bf, bv,
                                          q_bf, k_bf, vT_bf);

  // ---- pass 2: attention ----
  dim3 g2(S_ / 64, B_ * H_);
  attn_kernel<<<g2, 128, 0, stream>>>(q_bf, k_bf, vT_bf, o_bf);

  // ---- pass 3: output projection ----
  dim3 g3(M_ / 64, E_ / 128);
  out_proj_kernel<<<g3, 128, 0, stream>>>(o_bf, wo_bf, bo, out);
}
